// RNNModel_39668317946566
// MI455X (gfx1250) — matrix operand order repaired and checked by reading, not compiled, not run on hardware
//
#include <hip/hip_runtime.h>

// CDNA5 (gfx1250) fused RNN scan, v2:
//  - state transposed (h^T), W_hh resident in registers as WMMA A fragments
//  - W_hh rows PERMUTED across A fragments so each step's D tiles are, slot
//    for slot, the next step's B fragments: no cross-lane repack at all
//    (g(t,M) = 32*(t>>1) + 16*(M>>3) + 8*(t&1) + (M&7))
//  - input projection + biases folded into a 3rd K-chunk of the same WMMA
//  - native v_tanh_f32 (or branch-free exp2 fallback), no libm branches
//  - x load software-pipelined one step ahead, branch-free masking
// One wave owns a 16-batch tile; 128 single-wave blocks.

typedef __attribute__((ext_vector_type(16))) _Float16 v16h;
typedef __attribute__((ext_vector_type(8)))  float    v8f;

#define T_STEPS 2048
#define BATCH   2048
#define HDIM    64

__device__ __forceinline__ float fast_tanh(float v) {
#if __has_builtin(__builtin_amdgcn_tanhf)
    return __builtin_amdgcn_tanhf(v);            // V_TANH_F32 (TRANS)
#else
    // branch-free: tanh(v) = 1 - 2/(exp2(v*2*log2(e)) + 1)
    float e = __builtin_amdgcn_exp2f(v * 2.8853900817779268f);
    float r = __builtin_amdgcn_rcpf(e + 1.0f);
    return __builtin_fmaf(-2.0f, r, 1.0f);
#endif
}

__global__ __launch_bounds__(32) void rnn_scan_wmma(
    const float* __restrict__ x,     // [B, T, 2]
    const float* __restrict__ W_ih,  // [64, 2]
    const float* __restrict__ W_hh,  // [64, 64]
    const float* __restrict__ b_ih,  // [64]
    const float* __restrict__ b_hh,  // [64]
    const float* __restrict__ W_fc,  // [2, 64]
    const float* __restrict__ b_fc,  // [2]
    float* __restrict__ out)         // [B, 2]
{
    const int lane = threadIdx.x & 31;
    const int lm   = lane & 15;   // N (batch col) in B/D; M row slot in A
    const int lh   = lane >> 4;   // lane half
    const int base = blockIdx.x * 16;

    // ---- Preload W_hh as A fragments with the self-feeding row permutation.
    // A (16x32 f16): M = lane%16; half hh -> K = hh + 16*(lane/16)
    v16h A_hh[4][2];
#pragma unroll
    for (int t = 0; t < 4; ++t) {
        const int m = 32 * (t >> 1) + 16 * (lm >> 3) + 8 * (t & 1) + (lm & 7);
#pragma unroll
        for (int c = 0; c < 2; ++c) {
#pragma unroll
            for (int hh = 0; hh < 16; ++hh) {
                const int k = 32 * c + 16 * lh + hh;
                A_hh[t][c][hh] = (_Float16)W_hh[m * 64 + k];
            }
        }
    }
    // Augmented input chunk: K=0 -> W_ih[:,0], K=1 -> W_ih[:,1], K=2 -> bias
    v16h A_x[4];
#pragma unroll
    for (int t = 0; t < 4; ++t) {
        const int m = 32 * (t >> 1) + 16 * (lm >> 3) + 8 * (t & 1) + (lm & 7);
        v16h a;
#pragma unroll
        for (int e = 0; e < 16; ++e) a[e] = (_Float16)0.0f;
        if (lh == 0) {  // lanes>=16 map to K=16..18 which are zero rows
            a[0] = (_Float16)W_ih[m * 2 + 0];
            a[1] = (_Float16)W_ih[m * 2 + 1];
            a[2] = (_Float16)(b_ih[m] + b_hh[m]);
        }
        A_x[t] = a;
    }

    // ---- State h^T as two B fragments (K=0..31 / K=32..63), init zero.
    v16h Hf0, Hf1;
#pragma unroll
    for (int e = 0; e < 16; ++e) { Hf0[e] = (_Float16)0.0f; Hf1[e] = (_Float16)0.0f; }

    // x B-fragment: rows K=0 -> x0, K=1 -> x1, K=2 -> 1.0; rest permanently 0.
    v16h Xf;
#pragma unroll
    for (int e = 0; e < 16; ++e) Xf[e] = (_Float16)0.0f;
    const float xmask = (lh == 0) ? 1.0f : 0.0f;
    Xf[2] = (_Float16)xmask;

    const float2* __restrict__ xrow =
        (const float2*)(x + (size_t)(base + lm) * T_STEPS * 2);

    float2 cx = xrow[0];   // software pipeline: x for step t

    // ---------------- Sequential scan over time --------------------------
#pragma unroll 1
    for (int t = 0; t < T_STEPS; ++t) {
        Xf[0] = (_Float16)(cx.x * xmask);
        Xf[1] = (_Float16)(cx.y * xmask);

        // issue next step's x load early; wait overlaps the WMMAs
        if (t + 1 < T_STEPS) cx = xrow[t + 1];

        // G = [W_hh | W_ih | bias] @ [h^T ; x^T ; 1] : 4 M-tiles x 3 K-chunks
        v8f acc[4];
#pragma unroll
        for (int tt = 0; tt < 4; ++tt) {
            v8f c0;
#pragma unroll
            for (int j = 0; j < 8; ++j) c0[j] = 0.0f;
            c0 = __builtin_amdgcn_wmma_f32_16x16x32_f16(false, A_x[tt],     false, Xf,  (short)0, c0, false, false);
            c0 = __builtin_amdgcn_wmma_f32_16x16x32_f16(false, A_hh[tt][0], false, Hf0, (short)0, c0, false, false);
            c0 = __builtin_amdgcn_wmma_f32_16x16x32_f16(false, A_hh[tt][1], false, Hf1, (short)0, c0, false, false);
            acc[tt] = c0;
        }

        // tanh + repack: thanks to the row permutation, D tile t feeds
        // B fragment (t>>1), halves 8*(t&1)+r, same lane. Pure cvt/pack.
#pragma unroll
        for (int r = 0; r < 8; ++r) {
            Hf0[r]     = (_Float16)fast_tanh(acc[0][r]);
            Hf0[r + 8] = (_Float16)fast_tanh(acc[1][r]);
            Hf1[r]     = (_Float16)fast_tanh(acc[2][r]);
            Hf1[r + 8] = (_Float16)fast_tanh(acc[3][r]);
        }
    }

    // ---------------- FC head on h_T (tiny, via LDS) ---------------------
    // B slot (c, h, lh) holds state row k = 32c + 16*lh + h (canonical).
    __shared__ float lds_h[HDIM * 16];  // h^T[k][n]
#pragma unroll
    for (int c = 0; c < 2; ++c) {
        const v16h hf = (c == 0) ? Hf0 : Hf1;
#pragma unroll
        for (int h = 0; h < 16; ++h) {
            const int k = 32 * c + 16 * lh + h;
            lds_h[k * 16 + lm] = (float)hf[h];
        }
    }
    __syncthreads();  // single-wave WG: NOP per ISA, kept for safety

    {
        const int o = lh;   // output channel 0/1
        const int b = lm;   // batch within tile
        float sum = b_fc[o];
#pragma unroll
        for (int k = 0; k < HDIM; ++k)
            sum += lds_h[k * 16 + b] * W_fc[o * 64 + k];
        out[(size_t)(base + b) * 2 + o] = sum;
    }
}

extern "C" void kernel_launch(void* const* d_in, const int* in_sizes, int n_in,
                              void* d_out, int out_size, void* d_ws, size_t ws_size,
                              hipStream_t stream) {
    const float* x    = (const float*)d_in[0];
    const float* W_ih = (const float*)d_in[1];
    const float* W_hh = (const float*)d_in[2];
    const float* b_ih = (const float*)d_in[3];
    const float* b_hh = (const float*)d_in[4];
    const float* W_fc = (const float*)d_in[5];
    const float* b_fc = (const float*)d_in[6];

    rnn_scan_wmma<<<BATCH / 16, 32, 0, stream>>>(
        x, W_ih, W_hh, b_ih, b_hh, W_fc, b_fc, (float*)d_out);
}
